// TrackerV3_73254962201197
// MI455X (gfx1250) — compile-verified, matrix operand order checked by reading, not run on hardware
//
#include <hip/hip_runtime.h>

typedef __attribute__((ext_vector_type(16))) _Float16 v16h;
typedef __attribute__((ext_vector_type(8)))  float    v8f;

#define NPTS   512
#define SFRM   16
#define CCH    128
#define HGT    64
#define WID    64
#define HID    384
#define TDIM_C 452       // logical input width
#define TDIM_P 512       // padded h_in stride (K for input GEMM)
#define NFLOW  192       // padded flow-head width (cols 130.. are zero)
#define ROWS   8192      // NPTS*SFRM
#define ITERS  4
#define KSTEP  64

__device__ __forceinline__ float gelu_f(float x) {
  return 0.5f * x * (1.0f + erff(x * 0.7071067811865476f));
}

// ---------------------------------------------------------------------------
// WMMA GEMM:  C[M,N] = act(A[M,K] @ B[K,N] + bias) (+ Res)
// Block = 128 threads = 4 waves; block tile = 16(M) x 64(N); wave w owns
// cols [n0 + 16w, +16).  A/B fp32 tiles are staged into LDS as f16.
// Fast path: unguarded float4 loads.  Edge path: unconditional loads from
// clamped addresses + select-to-zero.  All pipeline shapes are padded so the
// fast path is the only one that executes.  Fragments are read from LDS as
// contiguous 16B chunks (ds_load_b128) in the exact v_wmma_f32_16x16x32_f16
// register layout.  act: 0=none, 1=gelu (applied before residual).
// ---------------------------------------------------------------------------
__global__ __launch_bounds__(128) void trk_gemm_wmma(
    const float* __restrict__ A, const float* __restrict__ B,
    const float* __restrict__ bias, const float* __restrict__ Res,
    float* __restrict__ C, int M, int N, int K, int act)
{
  __shared__ _Float16 Asm[16 * KSTEP];   // [m][k]
  __shared__ _Float16 Bsm[64 * KSTEP];   // [n][k]  (transposed tile)

  const int tid  = threadIdx.x;
  const int wave = tid >> 5;
  const int lane = tid & 31;
  const int m0 = blockIdx.y * 16;
  const int n0 = blockIdx.x * 64;
  const int r  = lane & 15;           // fragment row (A) / col (B)
  const int kg = (lane >> 4) * 8;     // K sub-group base within 16
  const bool n4 = ((N & 3) == 0);

  v8f acc = {};

  for (int k0 = 0; k0 < K; k0 += KSTEP) {
    // ---- stage A tile: 16 rows x 64 k-cols (8 consecutive floats/thread) ----
    {
      int idx = tid * 8;               // [0, 1024)
      int row = idx >> 6;              // /KSTEP
      int col = idx & (KSTEP - 1);
      int gr  = m0 + row;
      _Float16* dst = Asm + row * KSTEP + col;
      if (gr < M && (k0 + col + 8) <= K) {
        const float* ap = A + (size_t)gr * K + k0 + col;
        float4 v0 = *(const float4*)(ap);
        float4 v1 = *(const float4*)(ap + 4);
        dst[0] = (_Float16)v0.x; dst[1] = (_Float16)v0.y;
        dst[2] = (_Float16)v0.z; dst[3] = (_Float16)v0.w;
        dst[4] = (_Float16)v1.x; dst[5] = (_Float16)v1.y;
        dst[6] = (_Float16)v1.z; dst[7] = (_Float16)v1.w;
      } else {
        // unconditional clamped loads, then select-to-zero
        int grc = min(gr, M - 1);
        const float* ap = A + (size_t)grc * K;
        float tmp[8];
        #pragma unroll
        for (int u = 0; u < 8; ++u) {
          int kc = min(k0 + col + u, K - 1);
          tmp[u] = ap[kc];
        }
        #pragma unroll
        for (int u = 0; u < 8; ++u) {
          int k = k0 + col + u;
          dst[u] = (gr < M && k < K) ? (_Float16)tmp[u] : (_Float16)0.f;
        }
      }
    }
    // ---- stage B tile: 64 k-rows x 64 n-cols, store transposed [n][k] ----
    #pragma unroll
    for (int rep = 0; rep < 8; ++rep) {
      int idx  = rep * 512 + tid * 4;  // [0, 4096)
      int krow = idx >> 6;
      int ncol = idx & 63;
      int gk = k0 + krow;
      int gn = n0 + ncol;
      _Float16 h0, h1, h2, h3;
      if (n4 && gk < K && (gn + 4) <= N) {
        float4 v = *(const float4*)(B + (size_t)gk * N + gn);
        h0 = (_Float16)v.x; h1 = (_Float16)v.y;
        h2 = (_Float16)v.z; h3 = (_Float16)v.w;
      } else {
        // unconditional clamped loads, then select-to-zero
        int gkc = min(gk, K - 1);
        const float* bp = B + (size_t)gkc * N;
        int c0 = min(gn + 0, N - 1), c1 = min(gn + 1, N - 1);
        int c2 = min(gn + 2, N - 1), c3 = min(gn + 3, N - 1);
        float v0 = bp[c0], v1 = bp[c1], v2 = bp[c2], v3 = bp[c3];
        bool kv = (gk < K);
        h0 = (kv && gn + 0 < N) ? (_Float16)v0 : (_Float16)0.f;
        h1 = (kv && gn + 1 < N) ? (_Float16)v1 : (_Float16)0.f;
        h2 = (kv && gn + 2 < N) ? (_Float16)v2 : (_Float16)0.f;
        h3 = (kv && gn + 3 < N) ? (_Float16)v3 : (_Float16)0.f;
      }
      Bsm[(ncol + 0) * KSTEP + krow] = h0;
      Bsm[(ncol + 1) * KSTEP + krow] = h1;
      Bsm[(ncol + 2) * KSTEP + krow] = h2;
      Bsm[(ncol + 3) * KSTEP + krow] = h3;
    }
    __syncthreads();

    // ---- two 16x16x32 WMMAs from LDS fragments ----
    const _Float16* ar = Asm + r * KSTEP;
    const _Float16* br = Bsm + (wave * 16 + r) * KSTEP;
    #pragma unroll
    for (int ks = 0; ks < KSTEP; ks += 32) {
      v16h af, bf;
      #pragma unroll
      for (int i = 0; i < 8; ++i) {
        af[i]     = ar[ks + kg + i];
        af[i + 8] = ar[ks + 16 + kg + i];
        bf[i]     = br[ks + kg + i];
        bf[i + 8] = br[ks + 16 + kg + i];
      }
      acc = __builtin_amdgcn_wmma_f32_16x16x32_f16(false, af, false, bf,
                                                   (short)0, acc, false, false);
    }
    __syncthreads();
  }

  // ---- epilogue ----
  const int bcol = n0 + wave * 16 + r;
  if (bcol < N) {
    const float bv = bias ? bias[bcol] : 0.f;
    #pragma unroll
    for (int j = 0; j < 8; ++j) {
      int cm = m0 + j + (lane >> 4) * 8;
      if (cm < M) {
        float v = acc[j] + bv;
        if (act == 1) v = gelu_f(v);
        if (Res) v += Res[(size_t)cm * N + bcol];
        C[(size_t)cm * N + bcol] = v;
      }
    }
  }
}

// ---------------------------------------------------------------------------
// LayerNorm, one wave per row.  X row stride ldx, Y row stride ldy.
// ---------------------------------------------------------------------------
__global__ void trk_layernorm(const float* __restrict__ X, int ldx,
                              const float* __restrict__ w, const float* __restrict__ b,
                              float* __restrict__ Y, int ldy, int M, int D)
{
  int row  = blockIdx.x * (blockDim.x >> 5) + (threadIdx.x >> 5);
  int lane = threadIdx.x & 31;
  if (row >= M) return;
  const float* x = X + (size_t)row * ldx;
  float s = 0.f, s2 = 0.f;
  for (int i = lane; i < D; i += 32) { float v = x[i]; s += v; s2 += v * v; }
  for (int off = 16; off; off >>= 1) {
    s  += __shfl_xor(s,  off, 32);
    s2 += __shfl_xor(s2, off, 32);
  }
  float mean = s / (float)D;
  float var  = s2 / (float)D - mean * mean;
  float inv  = rsqrtf(var + 1e-5f);
  float* y = Y + (size_t)row * ldy;
  for (int i = lane; i < D; i += 32)
    y[i] = (x[i] - mean) * inv * w[i] + b[i];
}

// ---------------------------------------------------------------------------
// Attention: 512 seqs x 8 heads x 16 query positions, head_dim 48.
// One thread per (n, t, head).  QKV rows [8192 x 1152], O rows [8192 x 384].
// ---------------------------------------------------------------------------
__global__ void trk_attention(const float* __restrict__ QKV, float* __restrict__ O)
{
  int tid = blockIdx.x * blockDim.x + threadIdx.x;
  if (tid >= NPTS * SFRM * 8) return;
  int hd = tid & 7;
  int t  = (tid >> 3) & 15;
  int n  = tid >> 7;
  const float* base = QKV + (size_t)n * SFRM * 1152;
  const float* q = base + (size_t)t * 1152 + hd * 48;
  float scores[SFRM];
  float mx = -1e30f;
  for (int s = 0; s < SFRM; ++s) {
    const float* k = base + (size_t)s * 1152 + 384 + hd * 48;
    float d = 0.f;
    for (int c = 0; c < 48; ++c) d += q[c] * k[c];
    d *= 0.144337567297406441f; // 1/sqrt(48)
    scores[s] = d;
    mx = fmaxf(mx, d);
  }
  float den = 0.f;
  for (int s = 0; s < SFRM; ++s) { scores[s] = expf(scores[s] - mx); den += scores[s]; }
  float rden = 1.f / den;
  float* o = O + (size_t)(n * SFRM + t) * HID + hd * 48;
  for (int c = 0; c < 48; ++c) {
    float acc = 0.f;
    for (int s = 0; s < SFRM; ++s)
      acc += scores[s] * base[(size_t)s * 1152 + 768 + hd * 48 + c];
    o[c] = acc * rden;
  }
}

// ---------------------------------------------------------------------------
// 2x2 average pool over the last two dims.  SC = S*C fused.
// ---------------------------------------------------------------------------
__global__ void trk_avgpool2(const float* __restrict__ src, float* __restrict__ dst,
                             int SC, int Hin, int Win)
{
  int Ho = Hin >> 1, Wo = Win >> 1;
  int idx = blockIdx.x * blockDim.x + threadIdx.x;
  int total = SC * Ho * Wo;
  if (idx >= total) return;
  int x = idx % Wo;
  int y = (idx / Wo) % Ho;
  int sc = idx / (Wo * Ho);
  const float* p = src + ((size_t)sc * Hin + 2 * y) * Win + 2 * x;
  dst[idx] = 0.25f * (p[0] + p[1] + p[Win] + p[Win + 1]);
}

// coords[s][n][2] = xys[n]/8 for all s
__global__ void trk_init_coords(const float* __restrict__ xys, float* __restrict__ coords)
{
  int n = blockIdx.x * blockDim.x + threadIdx.x;
  if (n >= NPTS) return;
  float cx = xys[n * 2 + 0] * 0.125f;
  float cy = xys[n * 2 + 1] * 0.125f;
  for (int s = 0; s < SFRM; ++s) {
    coords[(s * NPTS + n) * 2 + 0] = cx;
    coords[(s * NPTS + n) * 2 + 1] = cy;
  }
}

// zero-padded weight copies (one-time)
__global__ void trk_pad_win(const float* __restrict__ w_in, float* __restrict__ w_in_pad)
{
  int idx = blockIdx.x * blockDim.x + threadIdx.x;
  if (idx >= TDIM_P * HID) return;
  int k = idx / HID, n2 = idx % HID;
  w_in_pad[idx] = (k < TDIM_C) ? w_in[(size_t)k * HID + n2] : 0.f;
}

__global__ void trk_pad_wflow(const float* __restrict__ w_flow, const float* __restrict__ b_flow,
                              float* __restrict__ w_flow_pad, float* __restrict__ b_flow_pad)
{
  int idx = blockIdx.x * blockDim.x + threadIdx.x;
  if (idx >= HID * NFLOW) return;
  int k = idx / NFLOW, n2 = idx % NFLOW;
  w_flow_pad[idx] = (n2 < 130) ? w_flow[(size_t)k * 130 + n2] : 0.f;
  if (idx < NFLOW) b_flow_pad[idx] = (idx < 130) ? b_flow[idx] : 0.f;
}

// query feature: bilinear sample of frame-0 feature map at coords[:,0];
// broadcast over S into tf[(n*16+s)*128+c]
__global__ void trk_query_feat(const float* __restrict__ fmaps,
                               const float* __restrict__ coords, float* __restrict__ tf)
{
  int idx = blockIdx.x * blockDim.x + threadIdx.x;
  if (idx >= NPTS * CCH) return;
  int c = idx & 127, n = idx >> 7;
  float cx = fminf(fmaxf(coords[n * 2 + 0], 0.f), (float)(WID - 1));
  float cy = fminf(fmaxf(coords[n * 2 + 1], 0.f), (float)(HGT - 1));
  int x0 = (int)floorf(cx), y0 = (int)floorf(cy);
  int x1 = min(x0 + 1, WID - 1), y1 = min(y0 + 1, HGT - 1);
  float wx = cx - (float)x0, wy = cy - (float)y0;
  const float* img = fmaps + (size_t)c * (HGT * WID); // frame 0
  float v00 = img[y0 * WID + x0], v01 = img[y0 * WID + x1];
  float v10 = img[y1 * WID + x0], v11 = img[y1 * WID + x1];
  float v = v00 * (1 - wx) * (1 - wy) + v01 * wx * (1 - wy)
          + v10 * (1 - wx) * wy + v11 * wx * wy;
  for (int s = 0; s < SFRM; ++s)
    tf[((size_t)n * SFRM + s) * CCH + c] = v;
}

// sampled positional embedding spe[n][d] (analytic sincos grid, bilinear at coords[:,0])
__global__ void trk_spe(const float* __restrict__ coords, float* __restrict__ spe)
{
  int idx = blockIdx.x * blockDim.x + threadIdx.x;
  if (idx >= NPTS * TDIM_C) return;
  int d = idx % TDIM_C, n = idx / TDIM_C;
  float cx = fminf(fmaxf(coords[n * 2 + 0], 0.f), (float)(WID - 1));
  float cy = fminf(fmaxf(coords[n * 2 + 1], 0.f), (float)(HGT - 1));
  int x0 = (int)floorf(cx), y0 = (int)floorf(cy);
  int x1 = min(x0 + 1, WID - 1), y1 = min(y0 + 1, HGT - 1);
  float wx = cx - (float)x0, wy = cy - (float)y0;
  auto pv = [&](int xi, int yi) -> float {
    int dd = d; float p;
    if (dd < 226) { p = (float)xi; } else { p = (float)yi; dd -= 226; }
    int f = (dd < 113) ? dd : dd - 113;
    float omega = powf(10000.f, -(float)f / 113.f);
    float a = p * omega;
    return (dd < 113) ? sinf(a) : cosf(a);
  };
  float v00 = pv(x0, y0), v01 = pv(x1, y0), v10 = pv(x0, y1), v11 = pv(x1, y1);
  spe[idx] = v00 * (1 - wx) * (1 - wy) + v01 * wx * (1 - wy)
           + v10 * (1 - wx) * wy + v11 * wx * wy;
}

// ---------------------------------------------------------------------------
// Correlation sampling.  Block per (n, s); 256 threads = 4 groups of 64
// (one per pyramid level).  Each group computes corr at the 8x8 patch of
// integer corner pixels needed by the 7x7 integer-offset bilinear samples,
// then threads t<49 interpolate from LDS.  h_in row stride = TDIM_P.
// ---------------------------------------------------------------------------
__global__ void trk_fcorr(const float* __restrict__ f0, const float* __restrict__ f1,
                          const float* __restrict__ f2, const float* __restrict__ f3,
                          const float* __restrict__ coords, const float* __restrict__ tf,
                          float* __restrict__ h_in)
{
  int n = blockIdx.x, s = blockIdx.y;
  int tid = threadIdx.x;
  __shared__ float tfs[CCH];
  __shared__ float cor[4][64];
  if (tid < CCH) tfs[tid] = tf[((size_t)n * SFRM + s) * CCH + tid];
  __syncthreads();

  int lvl = tid >> 6;
  int t   = tid & 63;
  const float* fm = (lvl == 0) ? f0 : (lvl == 1) ? f1 : (lvl == 2) ? f2 : f3;
  int Wl = WID >> lvl;
  float scale = 1.f / (float)(1 << lvl);
  float cx = coords[(s * NPTS + n) * 2 + 0] * scale;
  float cy = coords[(s * NPTS + n) * 2 + 1] * scale;
  int bx = (int)floorf(cx) - 3;
  int by = (int)floorf(cy) - 3;
  int ix = t & 7, iy = t >> 3;
  int gx = min(max(bx + ix, 0), Wl - 1);
  int gy = min(max(by + iy, 0), Wl - 1);
  int HW = Wl * Wl;
  const float* col = fm + (size_t)s * CCH * HW + gy * Wl + gx;
  float dacc = 0.f;
  for (int c = 0; c < CCH; ++c) dacc += tfs[c] * col[(size_t)c * HW];
  cor[lvl][iy * 8 + ix] = dacc * 0.08838834764831845f; // 1/sqrt(128)
  __syncthreads();

  if (t < 49) {
    int i = t / 7, j = t % 7; // i -> x offset index, j -> y offset index
    float x = fminf(fmaxf(cx + (float)(i - 3), 0.f), (float)(Wl - 1));
    float y = fminf(fmaxf(cy + (float)(j - 3), 0.f), (float)(Wl - 1));
    float x0v = (float)min(max(bx + i, 0), Wl - 1);
    float y0v = (float)min(max(by + j, 0), Wl - 1);
    float wx = x - x0v, wy = y - y0v;
    const float* cl = cor[lvl];
    float v00 = cl[j * 8 + i],       v01 = cl[j * 8 + i + 1];
    float v10 = cl[(j + 1) * 8 + i], v11 = cl[(j + 1) * 8 + i + 1];
    float val = v00 * (1 - wx) * (1 - wy) + v01 * wx * (1 - wy)
              + v10 * (1 - wx) * wy + v11 * wx * wy;
    h_in[((size_t)n * SFRM + s) * TDIM_P + 128 + lvl * 49 + t] = val;
  }
}

// ---------------------------------------------------------------------------
// Assemble h_in row: flow sincos embedding [0..127], track feats [324..451],
// zero pad [452..511], and add spe[n] to the logical row (corr segment
// written earlier by trk_fcorr).  Block per (n,s), 256 threads.
// ---------------------------------------------------------------------------
__global__ void trk_build_base(const float* __restrict__ coords, const float* __restrict__ tf,
                               const float* __restrict__ spe, float* __restrict__ h_in)
{
  int n = blockIdx.x, s = blockIdx.y;
  int t = threadIdx.x;
  size_t row = (size_t)n * SFRM + s;
  float fx = coords[(s * NPTS + n) * 2 + 0] - coords[n * 2 + 0];
  float fy = coords[(s * NPTS + n) * 2 + 1] - coords[n * 2 + 1];
  const float* sp = spe + (size_t)n * TDIM_C;
  if (t < 128) {
    float v;
    if (t < 64) { int j = t >> 1;        float a = fx * 31.25f * (float)j; v = (t & 1) ? cosf(a) : sinf(a); }
    else        { int j = (t - 64) >> 1; float a = fy * 31.25f * (float)j; v = (t & 1) ? cosf(a) : sinf(a); }
    h_in[row * TDIM_P + t] = v + sp[t];
  } else {
    int c = t - 128;
    h_in[row * TDIM_P + 324 + c] = tf[row * CCH + c] + sp[324 + c];
  }
  for (int k = t; k < 196; k += 256)
    h_in[row * TDIM_P + 128 + k] += sp[128 + k];
  // zero the K padding (big buffer is reused by qkv/mlp between iterations)
  if (t < TDIM_P - TDIM_C)
    h_in[row * TDIM_P + TDIM_C + t] = 0.f;
}

// coords update (+ coord_preds output, scaled by STRIDE); delta stride NFLOW
__global__ void trk_update_coords(const float* __restrict__ delta, float* __restrict__ coords,
                                  float* __restrict__ out, int iter)
{
  int idx = blockIdx.x * blockDim.x + threadIdx.x;
  if (idx >= SFRM * NPTS) return;
  int n = idx & (NPTS - 1), s = idx >> 9;
  float cx = coords[(s * NPTS + n) * 2 + 0];
  float cy = coords[(s * NPTS + n) * 2 + 1];
  if (s > 0) {
    cx += delta[((size_t)n * SFRM + s) * NFLOW + 0] * 0.1f;
    cy += delta[((size_t)n * SFRM + s) * NFLOW + 1] * 0.1f;
    coords[(s * NPTS + n) * 2 + 0] = cx;
    coords[(s * NPTS + n) * 2 + 1] = cy;
  }
  out[(size_t)iter * (SFRM * NPTS * 2) + (s * NPTS + n) * 2 + 0] = cx * 8.f;
  out[(size_t)iter * (SFRM * NPTS * 2) + (s * NPTS + n) * 2 + 1] = cy * 8.f;
}

// visibility head: sigmoid(tf @ vis_w + vis_b) -> out[65536 + s*512 + n]
__global__ void trk_vis(const float* __restrict__ tf, const float* __restrict__ vw,
                        const float* __restrict__ vb, float* __restrict__ out)
{
  int idx = blockIdx.x * blockDim.x + threadIdx.x;
  if (idx >= SFRM * NPTS) return;
  int n = idx & (NPTS - 1), s = idx >> 9;
  const float* x = tf + ((size_t)n * SFRM + s) * CCH;
  float d = vb[0];
  for (int c = 0; c < CCH; ++c) d += x[c] * vw[c];
  out[ITERS * SFRM * NPTS * 2 + s * NPTS + n] = 1.f / (1.f + expf(-d));
}

// ---------------------------------------------------------------------------
// Host side
// ---------------------------------------------------------------------------
#define CDIV(a, b) (((a) + (b) - 1) / (b))

static inline void launch_gemm(const float* A, const float* B, const float* bias,
                               const float* Res, float* C, int M, int N, int K,
                               int act, hipStream_t stream)
{
  dim3 grid(CDIV(N, 64), CDIV(M, 16));
  trk_gemm_wmma<<<grid, dim3(128), 0, stream>>>(A, B, bias, Res, C, M, N, K, act);
}

static inline void launch_ln(const float* X, int ldx, const float* w, const float* b,
                             float* Y, int ldy, int M, int D, hipStream_t stream)
{
  trk_layernorm<<<CDIV(M, 8), 256, 0, stream>>>(X, ldx, w, b, Y, ldy, M, D);
}

extern "C" void kernel_launch(void* const* d_in, const int* in_sizes, int n_in,
                              void* d_out, int out_size, void* d_ws, size_t ws_size,
                              hipStream_t stream)
{
  const float* xys     = (const float*)d_in[0];
  const float* fmaps   = (const float*)d_in[1];
  // d_in[2] = iters (python int, fixed = 4 in harness)
  const float* w_in    = (const float*)d_in[3];
  const float* b_in    = (const float*)d_in[4];
  const float* ln1_w   = (const float*)d_in[5];
  const float* ln1_b   = (const float*)d_in[6];
  const float* qkv_w   = (const float*)d_in[7];
  const float* qkv_b   = (const float*)d_in[8];
  const float* proj_w  = (const float*)d_in[9];
  const float* proj_b  = (const float*)d_in[10];
  const float* ln2_w   = (const float*)d_in[11];
  const float* ln2_b   = (const float*)d_in[12];
  const float* mlp_w1  = (const float*)d_in[13];
  const float* mlp_b1  = (const float*)d_in[14];
  const float* mlp_w2  = (const float*)d_in[15];
  const float* mlp_b2  = (const float*)d_in[16];
  const float* w_flow  = (const float*)d_in[17];
  const float* b_flow  = (const float*)d_in[18];
  const float* gn_w    = (const float*)d_in[19];
  const float* gn_b    = (const float*)d_in[20];
  const float* ffeat_w = (const float*)d_in[21];
  const float* ffeat_b = (const float*)d_in[22];
  const float* vis_w   = (const float*)d_in[23];
  const float* vis_b   = (const float*)d_in[24];
  float* out = (float*)d_out;

  // workspace layout (floats); total ~26M floats (~104 MB)
  float* ws      = (float*)d_ws;
  float* pyr1    = ws;                     // 16*128*32*32 = 2097152
  float* pyr2    = pyr1 + 2097152;         // 524288
  float* pyr3    = pyr2 + 524288;          // 131072
  float* coords  = pyr3 + 131072;          // 16*512*2 = 16384
  float* tf      = coords + 16384;         // 512*16*128 = 1048576
  float* spe     = tf + 1048576;           // 512*452 = 231424
  float* hbuf    = spe + 231424;           // 8192*384 = 3145728
  float* abuf    = hbuf + 3145728;         // 8192*384 = 3145728
  float* dbuf    = abuf + 3145728;         // 8192*192 = 1572864 (padded flow out)
  float* nrm     = dbuf + 1572864;         // 8192*128 = 1048576
  float* w_in_p  = nrm + 1048576;          // 512*384 = 196608 (zero-padded w_in)
  float* w_fl_p  = w_in_p + 196608;        // 384*192 = 73728 (zero-padded w_flow)
  float* b_fl_p  = w_fl_p + 73728;         // 192 (round up 256)
  float* big     = b_fl_p + 256;           // 8192*1536 = 12582912 (h_in / qkv / mlp hidden)

  // --- one-time setup ---
  {
    int t1 = 2048 * 32 * 32;
    trk_avgpool2<<<CDIV(t1, 256), 256, 0, stream>>>(fmaps, pyr1, 2048, 64, 64);
    int t2 = 2048 * 16 * 16;
    trk_avgpool2<<<CDIV(t2, 256), 256, 0, stream>>>(pyr1, pyr2, 2048, 32, 32);
    int t3 = 2048 * 8 * 8;
    trk_avgpool2<<<CDIV(t3, 256), 256, 0, stream>>>(pyr2, pyr3, 2048, 16, 16);
    trk_init_coords<<<CDIV(NPTS, 256), 256, 0, stream>>>(xys, coords);
    trk_query_feat<<<CDIV(NPTS * CCH, 256), 256, 0, stream>>>(fmaps, coords, tf);
    trk_spe<<<CDIV(NPTS * TDIM_C, 256), 256, 0, stream>>>(coords, spe);
    trk_pad_win<<<CDIV(TDIM_P * HID, 256), 256, 0, stream>>>(w_in, w_in_p);
    trk_pad_wflow<<<CDIV(HID * NFLOW, 256), 256, 0, stream>>>(w_flow, b_flow, w_fl_p, b_fl_p);
  }

  for (int iter = 0; iter < ITERS; ++iter) {
    // correlation sampling + input assembly into big (= h_in, stride TDIM_P)
    trk_fcorr<<<dim3(NPTS, SFRM), 256, 0, stream>>>(fmaps, pyr1, pyr2, pyr3,
                                                    coords, tf, big);
    trk_build_base<<<dim3(NPTS, SFRM), 256, 0, stream>>>(coords, tf, spe, big);

    // input projection: h = h_in @ w_in + b_in  (K padded to 512, all fast path)
    launch_gemm(big, w_in_p, b_in, nullptr, hbuf, ROWS, HID, TDIM_P, 0, stream);

    for (int l = 0; l < 6; ++l) {
      launch_ln(hbuf, HID, ln1_w + l * HID, ln1_b + l * HID, abuf, HID, ROWS, HID, stream);
      launch_gemm(abuf, qkv_w + (size_t)l * HID * 1152, qkv_b + l * 1152,
                  nullptr, big, ROWS, 1152, HID, 0, stream);
      trk_attention<<<CDIV(NPTS * SFRM * 8, 256), 256, 0, stream>>>(big, abuf);
      launch_gemm(abuf, proj_w + (size_t)l * HID * HID, proj_b + l * HID,
                  hbuf, hbuf, ROWS, HID, HID, 0, stream);
      launch_ln(hbuf, HID, ln2_w + l * HID, ln2_b + l * HID, abuf, HID, ROWS, HID, stream);
      launch_gemm(abuf, mlp_w1 + (size_t)l * HID * 1536, mlp_b1 + l * 1536,
                  nullptr, big, ROWS, 1536, HID, 1, stream);            // gelu
      launch_gemm(big, mlp_w2 + (size_t)l * 1536 * HID, mlp_b2 + l * HID,
                  hbuf, hbuf, ROWS, HID, 1536, 0, stream);
    }

    // flow head: delta = h @ w_flow + b_flow  (N padded to 192, all fast path)
    launch_gemm(hbuf, w_fl_p, b_fl_p, nullptr, dbuf, ROWS, NFLOW, HID, 0, stream);

    // coords update + coord_preds[iter]
    trk_update_coords<<<CDIV(SFRM * NPTS, 256), 256, 0, stream>>>(dbuf, coords, out, iter);

    // feature update: tf += gelu(LN(dfeats) @ ffeat_w + ffeat_b)
    launch_ln(dbuf + 2, NFLOW, gn_w, gn_b, nrm, CCH, ROWS, CCH, stream);
    launch_gemm(nrm, ffeat_w, ffeat_b, tf, tf, ROWS, CCH, CCH, 1, stream); // gelu then +res
  }

  trk_vis<<<CDIV(SFRM * NPTS, 256), 256, 0, stream>>>(tf, vis_w, vis_b, out);
}
